// OctreeConv_29265907155614
// MI455X (gfx1250) — matrix operand order repaired and checked by reading, not compiled, not run on hardware
//
#include <hip/hip_runtime.h>

#define C_IN  32
#define C_OUT 32
#define KDIM  27
#define NPTS  131072

typedef __attribute__((ext_vector_type(16))) __bf16 v16bf;
typedef __attribute__((ext_vector_type(8)))  __bf16 v8bf;
typedef __attribute__((ext_vector_type(8)))  float  v8f;
typedef __attribute__((ext_vector_type(4)))  unsigned int u32x4;

__device__ __forceinline__ unsigned short f2bf(float f) {
    unsigned u = __builtin_bit_cast(unsigned, f);
    u += 0x7fffu + ((u >> 16) & 1u);          // round-to-nearest-even
    return (unsigned short)(u >> 16);
}

// ---------------------------------------------------------------------------
// Kernel 1: swizzle weights into the CDNA5 16-bit A-matrix (16x32) lane layout
// per ISA 7.12.2, as BF16. Storage order: (mtile, ktile, lane, halfIdx).
// A[o][K] with K = channel c (0..31) inside k-tile kt (=kernel tap).
// ---------------------------------------------------------------------------
__global__ void prep_weights(const float* __restrict__ w,
                             unsigned short* __restrict__ aws) {
    int tid  = blockIdx.x * 256 + threadIdx.x;   // 0 .. 27647
    int i    = tid & 15;                         // half index within lane
    int lane = (tid >> 4) & 31;
    int rest = tid >> 9;
    int kt   = rest % KDIM;                      // kernel tap 0..26
    int mt   = rest / KDIM;                      // M-tile 0..1
    int v = i >> 1, p = i & 1;
    // lanes 0-15: VGPR0-3 -> K 0..7, VGPR4-7 -> K 16..23 ; lanes 16-31: +8
    int kk = (v < 4) ? (v * 2 + p) : (16 + (v - 4) * 2 + p);
    if (lane >= 16) kk += 8;
    int o = mt * 16 + (lane & 15);               // output channel (row M)
    aws[tid] = f2bf(w[o * (C_IN * KDIM) + kk * KDIM + kt]);
}

// ---------------------------------------------------------------------------
// Kernel 2: transpose data_in (32,N) f32 -> dataT (N,32) bf16 so one gathered
// column is a single contiguous 64-byte record.
// ---------------------------------------------------------------------------
__global__ void transpose_cast(const float* __restrict__ din,
                               unsigned short* __restrict__ dT) {
    int n = blockIdx.x * 256 + threadIdx.x;
    union { unsigned short s[32]; u32x4 q[4]; } t;
#pragma unroll
    for (int c = 0; c < C_IN; ++c)
        t.s[c] = f2bf(din[(size_t)c * NPTS + n]);   // coalesced across lanes
    u32x4* dst = (u32x4*)(dT + (size_t)n * C_IN);
#pragma unroll
    for (int q = 0; q < 4; ++q) dst[q] = t.q[q];
}

// ---------------------------------------------------------------------------
// Kernel 3: gather + GEMM. One wave32 handles TWO 16-point tiles x all 32
// output channels (4 accumulators). A (weights) read once from LDS per tap
// and reused by both point-tiles; B gathered via saddr-form 32-bit offsets.
// ---------------------------------------------------------------------------
__global__ void __launch_bounds__(256) octree_gemm(
    const unsigned short* __restrict__ dT,
    const int* __restrict__ octree,
    const unsigned short* __restrict__ aws,
    float* __restrict__ out) {

    __shared__ __align__(16) unsigned short lw[2 * KDIM * 32 * 16]; // 54 KB

    // cooperative stage of swizzled weights into LDS (16B vectors)
    {
        const u32x4* src = (const u32x4*)aws;
        u32x4* dst = (u32x4*)lw;
        for (int t = threadIdx.x; t < (2 * KDIM * 32 * 16) / 8; t += 256)
            dst[t] = src[t];
    }
    __syncthreads();

    const int lane = threadIdx.x & 31;
    const int wv   = threadIdx.x >> 5;       // 8 waves / block (wave32)
    const int nn   = lane & 15;              // B/D column (point within tile)
    const int hi   = lane >> 4;              // 0: K 0-7,16-23; 1: K 8-15,24-31
    const int n0   = (blockIdx.x * 8 + wv) * 32;   // two 16-pt tiles per wave

    v8f accA0 = {0.f,0.f,0.f,0.f,0.f,0.f,0.f,0.f};
    v8f accA1 = accA0, accB0 = accA0, accB1 = accA0;

    const int* orowA = octree + (size_t)(n0 + nn) * KDIM;
    const int* orowB = orowA + 16 * KDIM;
    const char* dTb  = (const char*)dT;            // uniform base (saddr)
    const unsigned hoff = (unsigned)hi * 16u;      // channel-half byte offset

    for (int k = 0; k < KDIM; ++k) {
        // 32-bit divergent byte offsets -> GVS (saddr) global_load_b128
        unsigned ja = ((unsigned)orowA[k] << 6) + hoff;
        unsigned jb = ((unsigned)orowB[k] << 6) + hoff;

        union { v16bf v; v8bf h[2]; } bA, bB, a0, a1;
        bA.h[0] = *(const v8bf*)(dTb + ja);
        bA.h[1] = *(const v8bf*)(dTb + ja + 32);
        bB.h[0] = *(const v8bf*)(dTb + jb);
        bB.h[1] = *(const v8bf*)(dTb + jb + 32);

        const v8bf* l0 = (const v8bf*)&lw[(k * 32 + lane) * 16];
        a0.h[0] = l0[0]; a0.h[1] = l0[1];
        const v8bf* l1 = (const v8bf*)&lw[((KDIM + k) * 32 + lane) * 16];
        a1.h[0] = l1[0]; a1.h[1] = l1[1];

        accA0 = __builtin_amdgcn_wmma_f32_16x16x32_bf16(
                    false, a0.v, false, bA.v, (short)0, accA0, false, false);
        accA1 = __builtin_amdgcn_wmma_f32_16x16x32_bf16(
                    false, a1.v, false, bA.v, (short)0, accA1, false, false);
        accB0 = __builtin_amdgcn_wmma_f32_16x16x32_bf16(
                    false, a0.v, false, bB.v, (short)0, accB0, false, false);
        accB1 = __builtin_amdgcn_wmma_f32_16x16x32_bf16(
                    false, a1.v, false, bB.v, (short)0, accB1, false, false);
    }

    // C/D layout: VGPR r, lane l -> M = r + 8*(l>=16), N = l&15
    const int mb = hi * 8;
    float* oA = out + (size_t)mb * NPTS + (n0 + nn);
    float* oB = oA + 16;
#pragma unroll
    for (int r = 0; r < 8; ++r) {
        oA[(size_t)r * NPTS]        = accA0[r];   // channels  0..15, tile A
        oA[(size_t)(16 + r) * NPTS] = accA1[r];   // channels 16..31, tile A
        oB[(size_t)r * NPTS]        = accB0[r];   // channels  0..15, tile B
        oB[(size_t)(16 + r) * NPTS] = accB1[r];   // channels 16..31, tile B
    }
}

// ---------------------------------------------------------------------------
extern "C" void kernel_launch(void* const* d_in, const int* in_sizes, int n_in,
                              void* d_out, int out_size, void* d_ws, size_t ws_size,
                              hipStream_t stream) {
    (void)in_sizes; (void)n_in; (void)out_size; (void)ws_size;
    const float* data_in = (const float*)d_in[0];
    const float* weights = (const float*)d_in[1];
    const int*   octree  = (const int*)d_in[2];
    float* out = (float*)d_out;

    unsigned short* aws = (unsigned short*)d_ws;              // 55296 B
    unsigned short* dT  = aws + 2 * KDIM * 32 * 16;           // 8 MB bf16 (N,32)

    prep_weights  <<<(2 * KDIM * 32 * 16) / 256, 256, 0, stream>>>(weights, aws);
    transpose_cast<<<NPTS / 256,                 256, 0, stream>>>(data_in, dT);
    octree_gemm   <<<NPTS / (32 * 8),            256, 0, stream>>>(dT, octree, aws, out);
}
